// CausalSelfAttention_10161892622754
// MI455X (gfx1250) — matrix operand order repaired and checked by reading, not compiled, not run on hardware
//
#include <hip/hip_runtime.h>

typedef __attribute__((ext_vector_type(16))) __bf16 v16bf;
typedef __attribute__((ext_vector_type(8)))  __bf16 v8bf;
typedef __attribute__((ext_vector_type(8)))  float  v8f;

#define BATCH 4
#define TDIM  2048
#define CDIM  1024
#define NHEAD 16
#define HDIM  64
#define MROWS (BATCH * TDIM)   // 8192
#define N3    (3 * CDIM)       // 3072

// ---------------------------------------------------------------------------
// fp32 -> bf16 straight convert
__global__ __launch_bounds__(256) void cvt_bf16_kernel(const float* __restrict__ in,
                                                       __bf16* __restrict__ out, int n) {
    int i = blockIdx.x * 256 + threadIdx.x;
    if (i < n) out[i] = (__bf16)in[i];
}

// W[K][N] fp32 -> Wt[N][K] bf16, LDS-tiled 32x32 so both sides are coalesced.
// grid = (N/32, K/32), block = 256 (32x8)
__global__ __launch_bounds__(256) void transpose_bf16_kernel(const float* __restrict__ W,
                                                             __bf16* __restrict__ Wt,
                                                             int K, int N) {
    __shared__ __bf16 tile[32][33];   // +1 pad to dodge bank conflicts
    const int tx = threadIdx.x & 31;
    const int ty = threadIdx.x >> 5;  // 0..7
    const int n0 = blockIdx.x * 32;
    const int k0 = blockIdx.y * 32;
#pragma unroll
    for (int i = 0; i < 32; i += 8)
        tile[ty + i][tx] = (__bf16)W[(size_t)(k0 + ty + i) * N + n0 + tx];
    __syncthreads();
#pragma unroll
    for (int i = 0; i < 32; i += 8)
        Wt[(size_t)(n0 + ty + i) * K + k0 + tx] = tile[tx][ty + i];
}

// ---------------------------------------------------------------------------
// Generic bf16 WMMA GEMM:  OUT[M][N] = A[M][K] * Bt[N][K]^T + bias
// Block = 256 threads (8 waves), block tile 64x256, wave tile 32x64.
template <bool OUTF32>
__global__ __launch_bounds__(256) void gemm_bf16_wmma_kernel(
    const __bf16* __restrict__ A, const __bf16* __restrict__ Bt,
    const float* __restrict__ bias, void* __restrict__ outp,
    int M, int N, int K) {
    const int lane = threadIdx.x & 31;
    const int wave = threadIdx.x >> 5;
    const int g    = lane >> 4;    // half-wave group
    const int ln   = lane & 15;
    const int wm = blockIdx.y * 64  + (wave >> 2) * 32;   // wave row base
    const int wn = blockIdx.x * 256 + (wave & 3) * 64;    // wave col base

    v8f acc[2][4] = {};

    for (int k0 = 0; k0 < K; k0 += 32) {
        v16bf a[2], b[4];
        // A fragments: row = lane&15, k = 16g + [0..15] (contiguous, matches B)
#pragma unroll
        for (int i = 0; i < 2; ++i) {
            const __bf16* ap = A + (size_t)(wm + 16 * i + ln) * K + k0;
            v8bf lo = *(const v8bf*)(ap + 16 * g);
            v8bf hi = *(const v8bf*)(ap + 16 * g + 8);
#pragma unroll
            for (int e = 0; e < 8; ++e) { a[i][e] = lo[e]; a[i][8 + e] = hi[e]; }
        }
        // B fragments (contiguous k, same order as A): n = lane&15, k = 16g + i  -> contiguous in Bt row
#pragma unroll
        for (int c = 0; c < 4; ++c)
            b[c] = *(const v16bf*)(Bt + (size_t)(wn + 16 * c + ln) * K + k0 + 16 * g);

#pragma unroll
        for (int i = 0; i < 2; ++i)
#pragma unroll
            for (int c = 0; c < 4; ++c)
                acc[i][c] = __builtin_amdgcn_wmma_f32_16x16x32_bf16(
                    false, a[i], false, b[c], (short)0, acc[i][c], false, false);
    }

    // Epilogue: C-layout element (row = wm + 16i + r + 8g, col = wn + 16c + ln)
    float* outF  = (float*)outp;
    __bf16* outB = (__bf16*)outp;
#pragma unroll
    for (int i = 0; i < 2; ++i)
#pragma unroll
        for (int c = 0; c < 4; ++c) {
            const int col = wn + 16 * c + ln;
            const float bv = bias[col];
            const size_t base = (size_t)(wm + 16 * i + 8 * g) * N + col;
#pragma unroll
            for (int r = 0; r < 8; ++r) {
                float v = acc[i][c][r] + bv;
                if (OUTF32) outF[base + (size_t)r * N] = v;
                else        outB[base + (size_t)r * N] = (__bf16)v;
            }
        }
}

// ---------------------------------------------------------------------------
// qkv[M][3C] bf16 -> Q[BH][T][HD] (pre-scaled), K[BH][T][HD], Vt[BH][HD][T]
__global__ __launch_bounds__(256) void reshape_qkv_kernel(const __bf16* __restrict__ qkv,
                                                          __bf16* __restrict__ Q,
                                                          __bf16* __restrict__ Kc,
                                                          __bf16* __restrict__ Vt) {
    int idx = blockIdx.x * 256 + threadIdx.x;
    if (idx >= MROWS * N3) return;
    int n = idx % N3;
    int m = idx / N3;
    int b = m / TDIM, t = m % TDIM;
    int which = n >> 10;           // 0=Q 1=K 2=V
    int nn = n & (CDIM - 1);
    int h = nn >> 6, d = nn & 63;
    size_t bh = (size_t)(b * NHEAD + h);
    __bf16 v = qkv[(size_t)m * N3 + n];
    if (which == 0)      Q [(bh * TDIM + t) * HDIM + d] = (__bf16)((float)v * 0.125f); // 1/sqrt(64)
    else if (which == 1) Kc[(bh * TDIM + t) * HDIM + d] = v;
    else                 Vt[(bh * HDIM + d) * TDIM + t] = v;
}

// ---------------------------------------------------------------------------
// Flash attention with transposed score tiles (S^T = K * Q^T) so softmax
// stats are per-lane scalars: 2 cross-lane shuffles per key tile instead of 64.
// block = 4 waves; each wave owns 16 query rows of one (b,h).
// grid.x = B*NH*(T/64)
__global__ __launch_bounds__(128) void attn_flash_wmma_kernel(
    const __bf16* __restrict__ Qb, const __bf16* __restrict__ Kb,
    const __bf16* __restrict__ Vt, __bf16* __restrict__ Y) {
    const int qblk = blockIdx.x % (TDIM / 64);
    const int bh   = blockIdx.x / (TDIM / 64);
    const int h = bh % NHEAD, b = bh / NHEAD;
    const int lane = threadIdx.x & 31;
    const int wave = threadIdx.x >> 5;
    const int g = lane >> 4, ln = lane & 15;
    const int qrow0 = qblk * 64 + wave * 16;
    const int query = qrow0 + ln;          // this lane's query row (for S^T stats)

    const __bf16* Qp = Qb + (size_t)bh * TDIM * HDIM;
    const __bf16* Kp = Kb + (size_t)bh * TDIM * HDIM;
    const __bf16* Vp = Vt + (size_t)bh * HDIM * TDIM;   // [HD][T]

    __shared__ alignas(128) __bf16 lds_p[4][16 * 64];   // per-wave P staging [query][key]
    __bf16* pbuf = lds_p[wave];

    // Q as B-fragments (n = query lane, k = hd contiguous): one 32B load each
    v16bf qB[2];
#pragma unroll
    for (int f = 0; f < 2; ++f)
        qB[f] = *(const v16bf*)(Qp + (size_t)(qrow0 + ln) * HDIM + 32 * f + 16 * g);

    v8f o[4] = {};                 // O tiles: M=query rows, N=hd (16c+ln)
    float mq = -1e30f, lq = 0.0f;  // per-query softmax state (lane-resident)

    for (int kb = 0; kb <= qrow0 + 15; kb += 64) {
        // S^T tiles: st[c] = K[kb+16c..][:] x Q^T  -> (key rows, query cols)
        v8f st[4] = {};
#pragma unroll
        for (int c = 0; c < 4; ++c)
#pragma unroll
            for (int f = 0; f < 2; ++f) {
                const __bf16* ap = Kp + (size_t)(kb + 16 * c + ln) * HDIM + 32 * f;
                v8bf lo = *(const v8bf*)(ap + 16 * g);
                v8bf hi = *(const v8bf*)(ap + 16 * g + 8);
                v16bf kA;
#pragma unroll
                for (int e = 0; e < 8; ++e) { kA[e] = lo[e]; kA[8 + e] = hi[e]; }
                st[c] = __builtin_amdgcn_wmma_f32_16x16x32_bf16(
                    false, kA, false, qB[f], (short)0, st[c], false, false);
            }

        // causal mask: element key = kb+16c+r+8g, col = this lane's query
        if (kb + 63 > qrow0) {
#pragma unroll
            for (int c = 0; c < 4; ++c)
#pragma unroll
                for (int r = 0; r < 8; ++r)
                    if (kb + 16 * c + r + 8 * g > query) st[c][r] = -1e30f;
        }

        // per-query max: in-lane reduce (31 VALU) + one cross-half shuffle
        float mnew = st[0][0];
#pragma unroll
        for (int c = 0; c < 4; ++c)
#pragma unroll
            for (int r = 0; r < 8; ++r) mnew = fmaxf(mnew, st[c][r]);
        mnew = fmaxf(mnew, __shfl_xor(mnew, 16, 32));

        const float mt = fmaxf(mq, mnew);
        const float alpha = __expf(mq - mt);
        mq = mt;

        float rs = 0.0f;
#pragma unroll
        for (int c = 0; c < 4; ++c)
#pragma unroll
            for (int r = 0; r < 8; ++r) {
                float p = __expf(st[c][r] - mt);
                st[c][r] = p;
                rs += p;
            }
        rs += __shfl_xor(rs, 16, 32);
        lq = lq * alpha + rs;

        // rescale O: broadcast alpha from the lane owning query (r+8g)
        float ar[8];
#pragma unroll
        for (int r = 0; r < 8; ++r) ar[r] = __shfl(alpha, r + 8 * g, 16);
#pragma unroll
        for (int c = 0; c < 4; ++c)
#pragma unroll
            for (int r = 0; r < 8; ++r) o[c][r] *= ar[r];

        // pack P^T tile -> LDS [query][key]: one b128 store per (c)
#pragma unroll
        for (int c = 0; c < 4; ++c) {
            v8bf pk;
#pragma unroll
            for (int r = 0; r < 8; ++r) pk[r] = (__bf16)st[c][r];
            *(v8bf*)(pbuf + (size_t)ln * 64 + 16 * c + 8 * g) = pk;
        }
        asm volatile("s_wait_dscnt 0" ::: "memory");

        // reload P as A-layout fragments (row = query = ln, contraction = keys)
        v16bf pf[2];
#pragma unroll
        for (int f = 0; f < 2; ++f) {
            const __bf16* pp = pbuf + (size_t)ln * 64 + 32 * f;
            v8bf lo = *(const v8bf*)(pp + 16 * g);
            v8bf hi = *(const v8bf*)(pp + 16 * g + 8);
#pragma unroll
            for (int e = 0; e < 8; ++e) { pf[f][e] = lo[e]; pf[f][8 + e] = hi[e]; }
        }

        // O += P * V   (V B-fragments contiguous from transposed layout)
#pragma unroll
        for (int c = 0; c < 4; ++c)
#pragma unroll
            for (int f = 0; f < 2; ++f) {
                v16bf vf = *(const v16bf*)(Vp + (size_t)(16 * c + ln) * TDIM + kb + 32 * f + 16 * g);
                o[c] = __builtin_amdgcn_wmma_f32_16x16x32_bf16(
                    false, pf[f], false, vf, (short)0, o[c], false, false);
            }
    }

    // finalize: O /= l  (broadcast per-row 1/l), write Y[b][t][h*64 + d]
    const float inv = 1.0f / lq;
    float invr[8];
#pragma unroll
    for (int r = 0; r < 8; ++r) invr[r] = __shfl(inv, r + 8 * g, 16);
#pragma unroll
    for (int c = 0; c < 4; ++c)
#pragma unroll
        for (int r = 0; r < 8; ++r) {
            int row = qrow0 + r + 8 * g;
            int col = h * HDIM + 16 * c + ln;
            Y[((size_t)b * TDIM + row) * CDIM + col] = (__bf16)(o[c][r] * invr[r]);
        }
}

// ---------------------------------------------------------------------------
extern "C" void kernel_launch(void* const* d_in, const int* in_sizes, int n_in,
                              void* d_out, int out_size, void* d_ws, size_t ws_size,
                              hipStream_t stream) {
    (void)in_sizes; (void)n_in; (void)out_size; (void)ws_size;
    const float* x     = (const float*)d_in[0];
    const float* Wqkv  = (const float*)d_in[1];
    const float* bqkv  = (const float*)d_in[2];
    const float* Wproj = (const float*)d_in[3];
    const float* bproj = (const float*)d_in[4];
    float* out = (float*)d_out;

    char* ws = (char*)d_ws;
    size_t off = 0;
    auto take = [&](size_t bytes) -> void* {
        off = (off + 255) & ~(size_t)255;
        void* p = ws + off;
        off += bytes;
        return p;
    };
    __bf16* xb     = (__bf16*)take((size_t)MROWS * CDIM * 2);   // 16 MB
    __bf16* wqkvT  = (__bf16*)take((size_t)N3 * CDIM * 2);      //  6 MB
    __bf16* wprojT = (__bf16*)take((size_t)CDIM * CDIM * 2);    //  2 MB
    __bf16* qkvb   = (__bf16*)take((size_t)MROWS * N3 * 2);     // 48 MB
    __bf16* Qb     = (__bf16*)take((size_t)MROWS * CDIM * 2);   // 16 MB
    __bf16* Kb     = (__bf16*)take((size_t)MROWS * CDIM * 2);   // 16 MB
    __bf16* Vt     = (__bf16*)take((size_t)MROWS * CDIM * 2);   // 16 MB
    __bf16* Yb     = (__bf16*)take((size_t)MROWS * CDIM * 2);   // 16 MB

    // 1) convert/transpose operands to bf16
    {
        int n = MROWS * CDIM;
        cvt_bf16_kernel<<<(n + 255) / 256, 256, 0, stream>>>(x, xb, n);
    }
    transpose_bf16_kernel<<<dim3(N3 / 32, CDIM / 32), 256, 0, stream>>>(Wqkv, wqkvT, CDIM, N3);
    transpose_bf16_kernel<<<dim3(CDIM / 32, CDIM / 32), 256, 0, stream>>>(Wproj, wprojT, CDIM, CDIM);

    // 2) qkv = x @ Wqkv + bqkv  (bf16 out)
    gemm_bf16_wmma_kernel<false><<<dim3(N3 / 256, MROWS / 64), 256, 0, stream>>>(
        xb, wqkvT, bqkv, qkvb, MROWS, N3, CDIM);

    // 3) split/scale/transpose into attention layouts
    {
        int n = MROWS * N3;
        reshape_qkv_kernel<<<(n + 255) / 256, 256, 0, stream>>>(qkvb, Qb, Kb, Vt);
    }

    // 4) flash attention
    attn_flash_wmma_kernel<<<BATCH * NHEAD * (TDIM / 64), 128, 0, stream>>>(Qb, Kb, Vt, Yb);

    // 5) out = Y @ Wproj + bproj  (fp32 out)
    gemm_bf16_wmma_kernel<true><<<dim3(CDIM / 256, MROWS / 64), 256, 0, stream>>>(
        Yb, wprojT, bproj, out, MROWS, CDIM, CDIM);
}
